// qNetwork_SNN_8323646620583
// MI455X (gfx1250) — matrix operand order repaired and checked by reading, not compiled, not run on hardware
//
#include <hip/hip_runtime.h>

typedef __attribute__((ext_vector_type(16))) _Float16 v16h;
typedef __attribute__((ext_vector_type(8)))  _Float16 v8h;
typedef __attribute__((ext_vector_type(8)))  float    v8f;
typedef __attribute__((ext_vector_type(4)))  unsigned int u32x4;
typedef __attribute__((ext_vector_type(8)))  int      i32x8;
typedef __attribute__((ext_vector_type(4)))  int      i32x4;

#define BETA   0.95f
#define THRESH 1.0f

#define MODE_STORE_CUR 0
#define MODE_LIF       1
#define MODE_LIF_OUT   2

#if __has_builtin(__builtin_amdgcn_tensor_load_to_lds) && \
    __has_builtin(__builtin_amdgcn_s_wait_tensorcnt)
#define USE_TDM 1
#else
#define USE_TDM 0
#endif

// ---------------- elementwise helpers ----------------

__global__ void cvt_f32_f16(const float* __restrict__ src,
                            _Float16* __restrict__ dst, int n) {
  int i = blockIdx.x * blockDim.x + threadIdx.x;
  if (i < n) dst[i] = (_Float16)src[i];
}

// Layer-1 LIF: cur1 is precomputed (x@W1^T+b1 is timestep-invariant)
__global__ void lif_elementwise(const float* __restrict__ cur,
                                float* __restrict__ mem,
                                _Float16* __restrict__ spk, int n) {
  int i = blockIdx.x * blockDim.x + threadIdx.x;
  if (i < n) {
    float m = mem[i];
    float reset = (m > THRESH) ? 1.0f : 0.0f;   // spike(mem_prev - thr)
    m = BETA * m + cur[i] - reset * THRESH;
    float s = (m > THRESH) ? 1.0f : 0.0f;       // spike(mem_new - thr)
    mem[i] = m;
    spk[i] = (_Float16)s;
  }
}

// ---------------- fused WMMA GEMM + LIF ----------------
// C[M,N] = A[M,K] @ W[N,K]^T (+bias), then per-element LIF epilogue.
// Block tile 256(M) x 128(N), 8 waves as 4Mx2N, wave tile 64x64
// (16 accumulators), K-step 32 via v_wmma_f32_16x16x32_f16.
// B (weight) tile staged into LDS by the Tensor Data Mover when available;
// the D# pad fields reproduce the 40-half padded LDS row stride.

template<int MODE>
__global__ __launch_bounds__(256) void gemm_lif(
    const _Float16* __restrict__ A,   // [M x K] f16 row-major (spikes / input)
    const _Float16* __restrict__ W,   // [N x K] f16 row-major (weights)
    const float*    __restrict__ bias,// [N]
    float*          __restrict__ mem, // [M x N] f32 membrane state
    _Float16*       __restrict__ spk, // [M x N] f16 spikes out (MODE_LIF)
    float*          __restrict__ outacc, // [M x N] f32 accumulate (MODE_LIF_OUT)
    float*          __restrict__ curout, // [M x N] f32 (MODE_STORE_CUR)
    int N, int K)
{
  // rows padded to 40 halfs (80B): 16B aligned, staggers banks, and equals
  // TDM pad: 16 DWORDs data + 4 DWORDs pad per tile row.
  __shared__ _Float16 As[256][40];
  __shared__ _Float16 Bs[128][40];

  const int tid   = threadIdx.x;
  const int lane  = tid & 31;
  const int wave  = tid >> 5;
  const int waveM = wave >> 1;        // 0..3 -> 64 rows each
  const int waveN = wave & 1;         // 0..1 -> 64 cols each

  const int blockMBase = blockIdx.y * 256;
  const int blockNBase = blockIdx.x * 128;

  const int laneRow = lane & 15;
  const int laneHi  = lane >> 4;      // 0: lanes 0-15, 1: lanes 16-31

  v8f zero = {};
  v8f acc[4][4];
#pragma unroll
  for (int mi = 0; mi < 4; ++mi)
#pragma unroll
    for (int ni = 0; ni < 4; ++ni) acc[mi][ni] = zero;

#if USE_TDM
  // ---- uniform TDM descriptor pieces (B tile: 32 x 128 elements, 2B each) ----
  const unsigned ldsBOff = (unsigned)(size_t)&Bs[0][0];  // flat low 32b == LDS offset
  const unsigned long long wBase =
      (unsigned long long)(size_t)(W + (size_t)blockNBase * K);
#endif

  for (int k0 = 0; k0 < K; k0 += 32) {
    __syncthreads();
    // ---- A tile: 256 rows x 32 halfs, one row per thread ----
    {
      const v8h* ga = (const v8h*)(A + (size_t)(blockMBase + tid) * K + k0);
      v8h a0 = ga[0], a1 = ga[1];
      *(v8h*)&As[tid][0] = a0;
      *(v8h*)&As[tid][8] = a1;
      v8h a2 = ga[2], a3 = ga[3];
      *(v8h*)&As[tid][16] = a2;
      *(v8h*)&As[tid][24] = a3;
    }
#if USE_TDM
    // ---- B tile via Tensor Data Mover (issued by wave 0 only) ----
    if (wave == 0) {
      const unsigned long long gaddr = wBase + (unsigned long long)k0 * 2ull;
      u32x4 g0;
      g0[0] = 1u;                                   // count=1, user descriptor
      g0[1] = ldsBOff;                              // lds_addr (bytes)
      g0[2] = (unsigned)(gaddr & 0xFFFFFFFFull);    // global_addr[31:0]
      g0[3] = (unsigned)((gaddr >> 32) & 0x01FFFFFFull) | (2u << 30); // type=2
      i32x8 g1;
      g1[0] = (1 << 16)        // data_size = 2 bytes
            | (1 << 20)        // pad_enable
            | (3 << 22)        // pad_interval: 16 DWORDs
            | (3 << 25);       // pad_amount: 4 DWORDs
      g1[1] = (int)((unsigned)(K & 0xFFFF) << 16);          // tensor_dim0 lo16
      g1[2] = (int)(((unsigned)K >> 16) & 0xFFFFu)
            | (int)((unsigned)(N & 0xFFFF) << 16);          // dim0 hi / dim1 lo
      g1[3] = (int)(((unsigned)N >> 16) & 0xFFFFu)
            | (int)(32u << 16);                             // dim1 hi / tile_dim0=32
      g1[4] = 128;                                          // tile_dim1=128, tile_dim2=0
      g1[5] = K;                                            // tensor_dim0_stride lo32
      g1[6] = 0;                                            // stride hi / dim1_stride lo
      g1[7] = 0;
      i32x4 gz = {0, 0, 0, 0};
#if defined(__clang_major__) && (__clang_major__ >= 23)
      i32x8 gz8 = {0, 0, 0, 0, 0, 0, 0, 0};
      __builtin_amdgcn_tensor_load_to_lds(g0, g1, gz, gz, gz8, 0);
#else
      __builtin_amdgcn_tensor_load_to_lds(g0, g1, gz, gz, 0);
#endif
      __builtin_amdgcn_s_wait_tensorcnt(0);
    }
#else
    // ---- B tile: 128 rows x 32 halfs, cooperative ----
    {
      const int bRow = tid >> 1;
      const int bCol = (tid & 1) * 16;
      const v8h* gb = (const v8h*)(W + (size_t)(blockNBase + bRow) * K + k0 + bCol);
      v8h b0 = gb[0], b1 = gb[1];
      *(v8h*)&Bs[bRow][bCol]     = b0;
      *(v8h*)&Bs[bRow][bCol + 8] = b1;
    }
#endif
    __syncthreads();

    // B fragments (32x16 KxN): lane holds column laneRow; halfs 0-7 at K = laneHi*8, halfs 8-15 at K = 16 + laneHi*8
    v16h bfrag[4];
#pragma unroll
    for (int ni = 0; ni < 4; ++ni) {
      const int br = waveN * 64 + ni * 16 + laneRow;
      v8h lo = *(const v8h*)&Bs[br][laneHi * 8];
      v8h hi = *(const v8h*)&Bs[br][16 + laneHi * 8];
      bfrag[ni] = __builtin_shufflevector(lo, hi,
          0,1,2,3,4,5,6,7,8,9,10,11,12,13,14,15);
    }

    // A fragments (16x32 MxK): lane holds row laneRow;
    // halfs 0-7 at K = laneHi*8, halfs 8-15 at K = 16 + laneHi*8 (ISA layout)
#pragma unroll
    for (int mi = 0; mi < 4; ++mi) {
      const int ar = waveM * 64 + mi * 16 + laneRow;
      v8h lo = *(const v8h*)&As[ar][laneHi * 8];
      v8h hi = *(const v8h*)&As[ar][16 + laneHi * 8];
      v16h afrag = __builtin_shufflevector(lo, hi,
          0,1,2,3,4,5,6,7,8,9,10,11,12,13,14,15);
#pragma unroll
      for (int ni = 0; ni < 4; ++ni) {
        acc[mi][ni] = __builtin_amdgcn_wmma_f32_16x16x32_f16(
            false, afrag, false, bfrag[ni],
            (short)0, acc[mi][ni], false, false);
      }
    }
  }

  // Epilogue: C/D layout -> VGPR r: M = r + 8*laneHi, N = laneRow
#pragma unroll
  for (int mi = 0; mi < 4; ++mi) {
#pragma unroll
    for (int ni = 0; ni < 4; ++ni) {
      const int gcol = blockNBase + waveN * 64 + ni * 16 + laneRow;
      const float bv = bias[gcol];
#pragma unroll
      for (int r = 0; r < 8; ++r) {
        const int grow = blockMBase + waveM * 64 + mi * 16 + laneHi * 8 + r;
        const size_t idx = (size_t)grow * (size_t)N + (size_t)gcol;
        float curv = acc[mi][ni][r] + bv;
        if (MODE == MODE_STORE_CUR) {
          curout[idx] = curv;
        } else {
          float m = mem[idx];
          float reset = (m > THRESH) ? 1.0f : 0.0f;
          m = BETA * m + curv - reset * THRESH;
          float s = (m > THRESH) ? 1.0f : 0.0f;
          mem[idx] = m;
          if (MODE == MODE_LIF) {
            spk[idx] = (_Float16)s;
          } else {
            outacc[idx] += s;   // stream-ordered timesteps: race-free
          }
        }
      }
    }
  }
}

// ---------------- host orchestration ----------------

extern "C" void kernel_launch(void* const* d_in, const int* in_sizes, int n_in,
                              void* d_out, int out_size, void* d_ws, size_t ws_size,
                              hipStream_t stream) {
  (void)in_sizes; (void)n_in; (void)out_size; (void)ws_size;

  const int B  = 2048, IN = 512, H = 2048, OUT = 512, T = 16;

  const float* x  = (const float*)d_in[0];
  const float* W1 = (const float*)d_in[1];  const float* b1 = (const float*)d_in[2];
  const float* W2 = (const float*)d_in[3];  const float* b2 = (const float*)d_in[4];
  const float* W3 = (const float*)d_in[5];  const float* b3 = (const float*)d_in[6];
  const float* W4 = (const float*)d_in[7];  const float* b4 = (const float*)d_in[8];
  const float* W5 = (const float*)d_in[9];  const float* b5 = (const float*)d_in[10];
  float* out = (float*)d_out;

  // ---- workspace layout (byte offsets, MB-aligned regions) ----
  char* w = (char*)d_ws;
  const size_t MB = 1024ull * 1024ull;
  _Float16* xh  = (_Float16*)(w + 0);          // 2 MB  (B*IN*2)
  _Float16* W1h = (_Float16*)(w + 2 * MB);     // 2 MB  (H*IN*2)
  _Float16* W2h = (_Float16*)(w + 4 * MB);     // 8 MB
  _Float16* W3h = (_Float16*)(w + 12 * MB);    // 8 MB
  _Float16* W4h = (_Float16*)(w + 20 * MB);    // 8 MB
  _Float16* W5h = (_Float16*)(w + 28 * MB);    // 2 MB  (OUT*H*2)
  float*    cur1 = (float*)(w + 30 * MB);      // 16 MB (B*H*4)
  float*    mem1 = (float*)(w + 46 * MB);      // 16 MB
  float*    mem2 = (float*)(w + 62 * MB);      // 16 MB
  float*    mem3 = (float*)(w + 78 * MB);      // 16 MB
  float*    mem4 = (float*)(w + 94 * MB);      // 16 MB
  float*    mem5 = (float*)(w + 110 * MB);     // 4 MB  (B*OUT*4)
  _Float16* s1  = (_Float16*)(w + 114 * MB);   // 8 MB  (B*H*2)
  _Float16* s2  = (_Float16*)(w + 122 * MB);   // 8 MB
  _Float16* s3  = (_Float16*)(w + 130 * MB);   // 8 MB
  _Float16* s4  = (_Float16*)(w + 138 * MB);   // 8 MB -> total 146 MB

  // ---- one-time per launch: convert precision, zero state ----
  const int TPB = 256;
  cvt_f32_f16<<<(B * IN  + TPB - 1) / TPB, TPB, 0, stream>>>(x,  xh,  B * IN);
  cvt_f32_f16<<<(H * IN  + TPB - 1) / TPB, TPB, 0, stream>>>(W1, W1h, H * IN);
  cvt_f32_f16<<<(H * H   + TPB - 1) / TPB, TPB, 0, stream>>>(W2, W2h, H * H);
  cvt_f32_f16<<<(H * H   + TPB - 1) / TPB, TPB, 0, stream>>>(W3, W3h, H * H);
  cvt_f32_f16<<<(H * H   + TPB - 1) / TPB, TPB, 0, stream>>>(W4, W4h, H * H);
  cvt_f32_f16<<<(OUT * H + TPB - 1) / TPB, TPB, 0, stream>>>(W5, W5h, OUT * H);

  // mem1..mem5 are contiguous [46MB, 114MB); zero them + the output accumulator
  hipMemsetAsync(w + 46 * MB, 0, 68 * MB, stream);
  hipMemsetAsync(out, 0, (size_t)B * OUT * sizeof(float), stream);

  // ---- cur1 = xh @ W1h^T + b1 (timestep-invariant) ----
  {
    dim3 grid(H / 128, B / 256);
    gemm_lif<MODE_STORE_CUR><<<grid, 256, 0, stream>>>(
        xh, W1h, b1, nullptr, nullptr, nullptr, cur1, H, IN);
  }

  // ---- timestep loop: LIF1 (elementwise) then 4 fused GEMM+LIF layers ----
  dim3 gridH(H / 128, B / 256);     // 16 x 8 blocks
  dim3 gridO(OUT / 128, B / 256);   //  4 x 8 blocks
  for (int t = 0; t < T; ++t) {
    lif_elementwise<<<(B * H + TPB - 1) / TPB, TPB, 0, stream>>>(cur1, mem1, s1, B * H);

    gemm_lif<MODE_LIF><<<gridH, 256, 0, stream>>>(
        s1, W2h, b2, mem2, s2, nullptr, nullptr, H, H);
    gemm_lif<MODE_LIF><<<gridH, 256, 0, stream>>>(
        s2, W3h, b3, mem3, s3, nullptr, nullptr, H, H);
    gemm_lif<MODE_LIF><<<gridH, 256, 0, stream>>>(
        s3, W4h, b4, mem4, s4, nullptr, nullptr, H, H);
    gemm_lif<MODE_LIF_OUT><<<gridO, 256, 0, stream>>>(
        s4, W5h, b5, mem5, nullptr, out, nullptr, OUT, H);
  }
}
